// ResNet_11373073400147
// MI455X (gfx1250) — matrix operand order repaired and checked by reading, not compiled, not run on hardware
//
#include <hip/hip_runtime.h>

// ---------------------------------------------------------------------------
// Sparse 3x3x3 conv ResNet block for MI455X (gfx1250), bf16 WMMA version.
//   layer0: relu(sum_k gather(x,idx_k,mask_k) @ W0[k] + b0)   -> h1 (bf16)
//   layer1: sum_k gather(h1,...) @ W1[k] + b1 + x             -> out (f32)
//
// v2: weights staged in LDS (2 phases x 140KB, conflict-free 80B row stride),
//     1024-thread blocks (32 waves, 32 point-tiles) amortize staging,
//     branch-free masked gather via an appended zero row.
// ---------------------------------------------------------------------------

typedef __bf16 bf16_t;
typedef __attribute__((ext_vector_type(16))) __bf16        v16bf;
typedef __attribute__((ext_vector_type(8)))  float         v8f;
typedef __attribute__((ext_vector_type(4)))  unsigned int  u32x4;

#define NPTS    400000
#define CCH     64
#define KOFF    27
#define NTILES  (NPTS / 16)            // 25000, exact
#define WPACK   (KOFF * 2 * CCH * 32)  // packed bf16 weight elements = 110592
#define PHASE0K 14
#define PHASE1K 13
#define WROW    40                     // padded halfs per co-row in LDS (80 B)
#define LDSH    (PHASE0K * 2 * CCH * WROW)  // 71680 halfs = 143360 B

union Frag {
  v16bf v;
  u32x4 q[2];
};

// f32 -> bf16 activation conversion; also zeroes the appended zero-row (row N)
// of both activation buffers (used to realize the neighbor mask branch-free).
__global__ void cvt_f32_bf16_kernel(const float* __restrict__ x,
                                    bf16_t* __restrict__ xh,
                                    bf16_t* __restrict__ h1, int n) {
  int i = blockIdx.x * blockDim.x + threadIdx.x;
  if (i < n) {
    xh[i] = (bf16_t)x[i];
  } else if (i < n + CCH) {
    xh[i] = (bf16_t)0.f;
    h1[i] = (bf16_t)0.f;
  }
}

// Pack W[k][ci][co] (f32) into compact bf16 B-fragment layout:
//   Wp[((k*2+kb)*64 + co)*32 + j] = W[k][kb*32 + j][co]
__global__ void pack_weights_kernel(const float* __restrict__ W0,
                                    const float* __restrict__ W1,
                                    bf16_t* __restrict__ W0p,
                                    bf16_t* __restrict__ W1p) {
  int p = blockIdx.x * blockDim.x + threadIdx.x;
  if (p >= WPACK) return;
  int j  = p & 31;
  int co = (p >> 5) & 63;
  int kb = (p >> 11) & 1;
  int k  = p >> 12;
  int src = k * (CCH * CCH) + (kb * 32 + j) * CCH + co;
  W0p[p] = (bf16_t)W0[src];
  W1p[p] = (bf16_t)W1[src];
}

template <bool FIRST>
__global__ __launch_bounds__(1024)
void sparse_conv_layer_kernel(const bf16_t* __restrict__ act,   // [N+1,64] bf16 (row N = zeros)
                              const bf16_t* __restrict__ wp,    // packed weights [WPACK]
                              const float*  __restrict__ bias,  // [64]
                              const int*    __restrict__ nidx,  // [N,27]
                              const unsigned char* __restrict__ nmask, // [N,27] bool
                              const float*  __restrict__ xres,  // residual (layer1)
                              bf16_t* __restrict__ out_bf,      // layer0 out [N+1,64]
                              float*  __restrict__ out_f)       // layer1 out [N,64]
{
  __shared__ __align__(16) unsigned short smem[LDSH];

  const int wave  = threadIdx.x >> 5;           // 0..31 within block
  const int tile  = blockIdx.x * 32 + wave;
  const bool active = tile < NTILES;
  const int lane  = threadIdx.x & 31;
  const int m     = lane & 15;                  // point-in-tile / B,C,D column
  const int half  = lane >> 4;                  // K-half / M-half of C,D
  const int nbase = tile * 16;
  const int n     = active ? (nbase + m) : 0;

  v8f acc[4];
  #pragma unroll
  for (int ct = 0; ct < 4; ++ct)
    acc[ct] = (v8f){0.f, 0.f, 0.f, 0.f, 0.f, 0.f, 0.f, 0.f};

  for (int phase = 0; phase < 2; ++phase) {
    const int kbeg = phase * PHASE0K;
    const int kcnt = phase ? PHASE1K : PHASE0K;

    // ---- stage this phase's weights into LDS (all 1024 threads) ----------
    __syncthreads();  // previous phase fully consumed
    {
      const int chunks = kcnt * 2 * CCH * 4;    // 16-byte chunks
      for (int c = threadIdx.x; c < chunks; c += 1024) {
        const int r  = c >> 2;                  // co-row within phase
        const int cj = c & 3;
        const u32x4* src = (const u32x4*)(wp + ((size_t)(kbeg * 2 * CCH) + r) * 32 + cj * 8);
        *(u32x4*)&smem[(size_t)r * WROW + cj * 8] = *src;
      }
    }
    __syncthreads();

    if (active) {
      for (int kk = 0; kk < kcnt; ++kk) {
        const int k   = kbeg + kk;
        const int idx = nidx[n * KOFF + k];
        const int mk  = nmask[n * KOFF + k];
        const int row = mk ? idx : NPTS;        // masked -> zero row, no branch

        // A fragments for both K-blocks: 4 x b128 in one clause.
        // Row chunks (8 bf16 each) needed by this lane: half, half+2, half+4, half+6.
        const u32x4* pa = (const u32x4*)(act + (size_t)row * CCH) + half;
        Frag a0, a1;
        a0.q[0] = pa[0];
        a0.q[1] = pa[2];
        a1.q[0] = pa[4];
        a1.q[1] = pa[6];

        #pragma unroll
        for (int kb = 0; kb < 2; ++kb) {
          const Frag& a = kb ? a1 : a0;
          const int bbase = ((kk * 2 + kb) * CCH) * WROW + half * 8;
          #pragma unroll
          for (int ct = 0; ct < 4; ++ct) {
            Frag b;
            const u32x4* pb = (const u32x4*)&smem[bbase + (ct * 16 + m) * WROW];
            b.q[0] = pb[0];
            b.q[1] = pb[2];
            acc[ct] = __builtin_amdgcn_wmma_f32_16x16x32_bf16(
                false, a.v, false, b.v, (short)0, acc[ct], false, false);
          }
        }
      }
    }
  }

  if (!active) return;

  // Epilogue straight from the C/D layout: VGPR v -> row nbase + v + 8*half,
  // column ct*16 + m.
  #pragma unroll
  for (int ct = 0; ct < 4; ++ct) {
    const int   co = ct * 16 + m;
    const float bv = bias[co];
    #pragma unroll
    for (int v = 0; v < 8; ++v) {
      const int row = nbase + v + half * 8;
      float val = acc[ct][v] + bv;
      if (FIRST) {
        val = val > 0.f ? val : 0.f;                 // ReLU
        out_bf[(size_t)row * CCH + co] = (bf16_t)val;
      } else {
        out_f[(size_t)row * CCH + co] =
            val + xres[(size_t)row * CCH + co];      // residual add
      }
    }
  }
}

extern "C" void kernel_launch(void* const* d_in, const int* in_sizes, int n_in,
                              void* d_out, int out_size, void* d_ws, size_t ws_size,
                              hipStream_t stream) {
  const float* x    = (const float*)d_in[0];
  const float* W0   = (const float*)d_in[1];
  const float* b0   = (const float*)d_in[2];
  const float* W1   = (const float*)d_in[3];
  const float* b1   = (const float*)d_in[4];
  const int*   nidx = (const int*)d_in[5];
  const unsigned char* nmask = (const unsigned char*)d_in[6];
  float* out = (float*)d_out;

  // Workspace layout (256B aligned):
  //   xh  : (N+1)*64 bf16  (51.2 MB, row N = zeros)
  //   h1  : (N+1)*64 bf16  (51.2 MB, row N = zeros)
  //   W0p, W1p : 110592 bf16 each
  char* ws = (char*)d_ws;
  size_t off = 0;
  auto take = [&](size_t bytes) {
    char* p = ws + off;
    off += (bytes + 255) & ~(size_t)255;
    return p;
  };
  bf16_t* xh  = (bf16_t*)take((size_t)(NPTS + 1) * CCH * sizeof(bf16_t));
  bf16_t* h1  = (bf16_t*)take((size_t)(NPTS + 1) * CCH * sizeof(bf16_t));
  bf16_t* W0p = (bf16_t*)take((size_t)WPACK * sizeof(bf16_t));
  bf16_t* W1p = (bf16_t*)take((size_t)WPACK * sizeof(bf16_t));

  const int nelem = NPTS * CCH;
  cvt_f32_bf16_kernel<<<(nelem + CCH + 255) / 256, 256, 0, stream>>>(x, xh, h1, nelem);
  pack_weights_kernel<<<(WPACK + 255) / 256, 256, 0, stream>>>(W0, W1, W0p, W1p);

  // 32 waves (tiles) per 1024-thread block; ceil(25000/32) = 782 blocks.
  dim3 grid((NTILES + 31) / 32), block(1024);
  sparse_conv_layer_kernel<true><<<grid, block, 0, stream>>>(
      xh, W0p, b0, nidx, nmask, nullptr, h1, nullptr);
  sparse_conv_layer_kernel<false><<<grid, block, 0, stream>>>(
      h1, W1p, b1, nidx, nmask, x, nullptr, out);
}
